// SequentialVQVAE_27315992003199
// MI455X (gfx1250) — matrix operand order repaired and checked by reading, not compiled, not run on hardware
//
#include <hip/hip_runtime.h>
#include <hip/hip_bf16.h>

// ---------------------------------------------------------------------------
// Sequential VQ-VAE forward for MI455X (gfx1250, wave32, WMMA).
// - All GEMMs via v_wmma_f32_16x16x32_f16 (f16-staged weights, f32 accum).
// - GRU scans exploit per-row independence: each block keeps its h-chunk in
//   LDS and loops over the sequential dim locally (no global sync).
// - Decoder stages its per-block gi tile into LDS once via
//   GLOBAL_LOAD_ASYNC_TO_LDS_B128 (+ s_wait_asynccnt), then 64 steps read LDS.
// ---------------------------------------------------------------------------

typedef __attribute__((ext_vector_type(16))) _Float16 v16h;
typedef __attribute__((ext_vector_type(8)))  float    v8f;

#define DEV __device__ __forceinline__

static constexpr int TT  = 64;    // T = SEQ-1
static constexpr int BB  = 256;   // batch
static constexpr int SEQ = 65;
static constexpr int ROWS = TT * BB;    // 16384
static constexpr int SS  = 8;     // codebooks
static constexpr int KK  = 64;    // codes per book
static constexpr int EE  = 16;    // code dim
static constexpr int SED = SS * EE;     // 128
static constexpr float DECAY = 0.99f;
static constexpr float EPSV  = 1e-5f;

// ---- WMMA fragment helpers (A/B: contiguous K per half-wave, same in both; C/D per ISA 7.12.2) ------------------

// A fragment (16x32, f16) from an f32 row-major matrix (global or LDS).
// lane L: row = row0 + (L&15); K block base (L<16 ? 0 : 16),
// VGPR i -> K = base+2i (16 contiguous K per lane).
DEV v16h a_frag_f32(const float* src, int ld, int row0, int k0, bool relu_in) {
  const int lane = threadIdx.x & 31;
  const int m  = row0 + (lane & 15);
  const int kb = k0 + ((lane >> 4) << 4);
  v16h a;
#pragma unroll
  for (int i = 0; i < 8; ++i) {
    const int k = kb + (i << 1);
    float x0 = src[(size_t)m * ld + k];
    float x1 = src[(size_t)m * ld + k + 1];
    if (relu_in) { x0 = fmaxf(x0, 0.f); x1 = fmaxf(x1, 0.f); }
    a[2*i]   = (_Float16)x0;
    a[2*i+1] = (_Float16)x1;
  }
  return a;
}

// B fragment (32x16, f16) from torch-layout weight W[N,K] (B = W^T).
// lane L: col n = col0 + (L&15); K block = k0 + (L<16 ? 0 : 16),
// 16 contiguous halfs along K -> one 32B-aligned vector load.
DEV v16h b_frag_f16(const _Float16* W, int ldw, int col0, int k0) {
  const int lane = threadIdx.x & 31;
  const int n  = col0 + (lane & 15);
  const int kb = k0 + ((lane >> 4) << 4);
  return *(const v16h*)(W + (size_t)n * ldw + kb);
}

DEV v8f wmma_f16(v16h a, v16h b, v8f c) {
  return __builtin_amdgcn_wmma_f32_16x16x32_f16(false, a, false, b,
                                                (short)0, c, false, false);
}

// C/D layout: lane L holds col n=(L&15); VGPR r holds row r + 8*(L>>4).
DEV void store_d_lds(float* dst, int ld, int row0, int col0, v8f acc) {
  const int lane = threadIdx.x & 31;
  const int n  = col0 + (lane & 15);
  const int m0 = row0 + ((lane >> 4) << 3);
#pragma unroll
  for (int r = 0; r < 8; ++r) dst[(m0 + r) * ld + n] = acc[r];
}

// ---- CDNA5 async global->LDS copy (ASYNCcnt path) --------------------------
DEV void async_g2l_b128(uint32_t lds_byte_addr, uint64_t global_addr) {
  asm volatile("global_load_async_to_lds_b128 %0, %1, off"
               :: "v"(lds_byte_addr), "v"(global_addr) : "memory");
}
DEV void wait_asynccnt0() {
  asm volatile("s_wait_asynccnt 0x0" ::: "memory");
}

// ---- generic GEMM:  Y[M,N] = act(X[M,K] @ W[N,K]^T + bias) -----------------
// Each wave computes a 16x32 output tile: one A fragment feeds two WMMAs.
__global__ void gemm_bias_kernel(const float* __restrict__ X,
                                 const _Float16* __restrict__ W,
                                 const float* __restrict__ bias,
                                 float* __restrict__ Y,
                                 int M, int N, int K, int relu) {
  const int wave = blockIdx.x * (blockDim.x >> 5) + (threadIdx.x >> 5);
  const int ntn  = N >> 5;                 // N is always a multiple of 32
  const int mt   = wave / ntn;
  const int nt   = wave - mt * ntn;
  if (mt * 16 >= M) return;                // wave-uniform: EXEC stays all-ones
  v8f acc0 = {}, acc1 = {};
  for (int k0 = 0; k0 < K; k0 += 32) {
    if (k0 + 32 < K)                       // -> global_prefetch_b8
      __builtin_prefetch(&X[(size_t)(mt * 16) * K + k0 + 32], 0, 1);
    v16h a  = a_frag_f32(X, K, mt * 16, k0, false);
    v16h b0 = b_frag_f16(W, K, nt * 32, k0);
    v16h b1 = b_frag_f16(W, K, nt * 32 + 16, k0);
    acc0 = wmma_f16(a, b0, acc0);
    acc1 = wmma_f16(a, b1, acc1);
  }
  const int lane = threadIdx.x & 31;
  const int m0 = mt * 16 + ((lane >> 4) << 3);
  const int n0 = nt * 32 + (lane & 15);
  const float bn0 = bias ? bias[n0] : 0.f;
  const float bn1 = bias ? bias[n0 + 16] : 0.f;
#pragma unroll
  for (int r = 0; r < 8; ++r) {
    float v0 = acc0[r] + bn0;
    float v1 = acc1[r] + bn1;
    if (relu) { v0 = fmaxf(v0, 0.f); v1 = fmaxf(v1, 0.f); }
    Y[(size_t)(m0 + r) * N + n0]      = v0;
    Y[(size_t)(m0 + r) * N + n0 + 16] = v1;
  }
}

// ---- tiny front-end MLPs -> xs[T*B,32] -------------------------------------
__global__ void embed_kernel(const float* __restrict__ actions,
                             const float* __restrict__ rewards,
                             const float* ae_w1, const float* ae_b1,
                             const float* ae_w2, const float* ae_b2,
                             const float* re_w1, const float* re_b1,
                             const float* re_w2, const float* re_b2,
                             float* __restrict__ xs) {
  int idx = blockIdx.x * blockDim.x + threadIdx.x;
  if (idx >= ROWS) return;
  const int t = idx >> 8, b = idx & 255;            // row = t*B + b
  const float a0 = actions[(size_t)(b * SEQ + t) * 2 + 0];
  const float a1 = actions[(size_t)(b * SEQ + t) * 2 + 1];
  const float rw = rewards[b * SEQ + t];
  float ha[16], hr[16];
#pragma unroll
  for (int i = 0; i < 16; ++i) {
    ha[i] = fmaxf(ae_w1[i*2] * a0 + ae_w1[i*2+1] * a1 + ae_b1[i], 0.f);
    hr[i] = fmaxf(re_w1[i] * rw + re_b1[i], 0.f);
  }
  float* xrow = xs + (size_t)idx * 32;
#pragma unroll
  for (int o = 0; o < 16; ++o) {
    float s0 = ae_b2[o], s1 = re_b2[o];
#pragma unroll
    for (int i = 0; i < 16; ++i) {
      s0 += ae_w2[o*16 + i] * ha[i];
      s1 += re_w2[o*16 + i] * hr[i];
    }
    xrow[o] = s0; xrow[16 + o] = s1;
  }
}

__global__ void tgt_kernel(const float* __restrict__ actions,
                           float* __restrict__ tgt, int n) {
  int i = blockIdx.x * blockDim.x + threadIdx.x;
  if (i < n) tgt[i] = actions[(size_t)i * 2 + 1];   // actions[:,:,1]
}

__global__ void f32_to_f16_kernel(const float* __restrict__ s,
                                  _Float16* __restrict__ d, int n) {
  int i = blockIdx.x * blockDim.x + threadIdx.x;
  if (i < n) d[i] = (_Float16)s[i];
}

__global__ void concat_kernel(const float* __restrict__ a32,
                              const float* __restrict__ b128,
                              float* __restrict__ cat, int n) {
  int i = blockIdx.x * blockDim.x + threadIdx.x;
  if (i >= n) return;
  int row = i / 160, c = i - row * 160;
  cat[i] = (c < 32) ? a32[(size_t)row * 32 + c]
                    : b128[(size_t)row * 128 + (c - 32)];
}

// ---- encoder GRU: each block owns 32 batch rows, loops t locally -----------
__global__ void enc_gru_kernel(const float* __restrict__ gi_all,  // [T*B,192] (+bih)
                               const _Float16* __restrict__ Whh,  // [192,64] f16
                               const float* __restrict__ bhh,
                               float* __restrict__ hs) {          // [T,B,64]
  __shared__ float h[32 * 65];
  __shared__ float gh[32 * 193];
  const int tid  = threadIdx.x;
  const int wave = tid >> 5;
  const int base = blockIdx.x * 32;
  for (int i = tid; i < 32 * 64; i += 256) h[(i >> 6) * 65 + (i & 63)] = 0.f;
  __syncthreads();
  for (int t = 0; t < TT; ++t) {
    for (int tile = wave; tile < 24; tile += 8) {       // 2 Mt x 12 Nt
      const int mt = tile & 1, nt = tile >> 1;
      v8f acc = {};
#pragma unroll
      for (int k0 = 0; k0 < 64; k0 += 32) {
        v16h a = a_frag_f32(h, 65, mt * 16, k0, false);
        v16h b = b_frag_f16(Whh, 64, nt * 16, k0);
        acc = wmma_f16(a, b, acc);
      }
      store_d_lds(gh, 193, mt * 16, nt * 16, acc);
    }
    __syncthreads();
    for (int i = tid; i < 32 * 64; i += 256) {
      const int row = i >> 6, j = i & 63;
      const float* gir = gi_all + (size_t)(t * BB + base + row) * 192;
      const float hr = gh[row * 193 + j]        + bhh[j];
      const float hz = gh[row * 193 + 64 + j]   + bhh[64 + j];
      const float hn = gh[row * 193 + 128 + j]  + bhh[128 + j];
      const float r  = 1.f / (1.f + __expf(-(gir[j] + hr)));
      const float z  = 1.f / (1.f + __expf(-(gir[64 + j] + hz)));
      const float nn = tanhf(gir[128 + j] + r * hn);
      const float h2 = (1.f - z) * nn + z * h[row * 65 + j];
      h[row * 65 + j] = h2;
      hs[(size_t)t * BB * 64 + (size_t)(base + row) * 64 + j] = h2;
    }
    __syncthreads();
  }
}

// ---- VQ quantizer with sequential EMA (one block per codebook s) -----------
__global__ void quantize_kernel(const float* __restrict__ z_seq,   // [T*B,128]
                                const float* __restrict__ emb0,    // [S,K,E]
                                const float* __restrict__ cs0,     // [S,K]
                                const float* __restrict__ w0,      // [S,K,E]
                                float* __restrict__ m_T,           // [B,T,SE]
                                float* __restrict__ m_seq,         // [T*B,SE]
                                int*   __restrict__ idx_out,       // [B,T,S]
                                float* __restrict__ vq_loss) {
  __shared__ float emb[KK * EE];
  __shared__ float ecs[KK];
  __shared__ float ew[KK * EE];
  __shared__ float nsum;
  __shared__ float red[256];
  const int s = blockIdx.x, tid = threadIdx.x, b = tid;
  for (int i = tid; i < KK * EE; i += 256) {
    emb[i] = emb0[s * KK * EE + i];
    ew[i]  = w0[s * KK * EE + i];
  }
  if (tid < KK) ecs[tid] = cs0[s * KK + tid];
  float lloss = 0.f;
  __syncthreads();
  for (int t = 0; t < TT; ++t) {
    float z[EE];
    const size_t zb = (size_t)(t * BB + b) * SED + s * EE;
#pragma unroll
    for (int e = 0; e < EE; ++e) z[e] = z_seq[zb + e];
    int best = 0; float bd = 3.4e38f;
    for (int k = 0; k < KK; ++k) {
      float d = 0.f;
#pragma unroll
      for (int e = 0; e < EE; ++e) {
        const float df = z[e] - emb[k * EE + e];
        d += df * df;
      }
      if (d < bd) { bd = d; best = k; }
    }
#pragma unroll
    for (int e = 0; e < EE; ++e) {
      const float q = emb[best * EE + e];
      const float df = q - z[e];
      lloss += df * df;
      m_T[(size_t)b * TT * SED + (size_t)t * SED + s * EE + e] = q;
      m_seq[zb + e] = q;
    }
    idx_out[(size_t)b * TT * SS + t * SS + s] = best;
    __syncthreads();
    // EMA: decay-scale, scatter via LDS float atomics (ds_add_f32), renorm.
    if (tid < KK) ecs[tid] *= DECAY;
    for (int i = tid; i < KK * EE; i += 256) ew[i] *= DECAY;
    __syncthreads();
    atomicAdd(&ecs[best], 1.f - DECAY);
#pragma unroll
    for (int e = 0; e < EE; ++e)
      atomicAdd(&ew[best * EE + e], (1.f - DECAY) * z[e]);
    __syncthreads();
    if (tid == 0) {
      float sn = 0.f;
      for (int k = 0; k < KK; ++k) sn += ecs[k];
      nsum = sn;
    }
    __syncthreads();
    for (int i = tid; i < KK * EE; i += 256) {
      const int k = i >> 4;
      const float csk = (ecs[k] + EPSV) / (nsum + KK * EPSV) * nsum;
      emb[i] = ew[i] / csk;
    }
    __syncthreads();
  }
  red[tid] = lloss;
  __syncthreads();
  for (int off = 128; off > 0; off >>= 1) {
    if (tid < off) red[tid] += red[tid + off];
    __syncthreads();
  }
  if (tid == 0)
    atomicAdd(vq_loss, red[0] * (1.f / (32768.f * 256.f)));  // /(B*S*E)/Bb
}

// ---- decoder GRU + logit head + masked BCE (the ~35-GFLOP hot loop) --------
// gi tile (32 rows x 192 f32 = 24 KB) staged once into LDS asynchronously;
// gh buffer is reused as the logits scratch (phases are barrier-disjoint).
__global__ void dec_gru_kernel(const float* __restrict__ gi,      // [T*B,192] (+bih)
                               const float* __restrict__ h0,      // [T*B,64]
                               const _Float16* __restrict__ Whh,  // [192,64] f16
                               const float* __restrict__ bhh,
                               const _Float16* __restrict__ adw1, // [64,64] f16
                               const float* __restrict__ adb1,
                               const float* __restrict__ adw2,    // [64]
                               const float* __restrict__ adb2,
                               const float* __restrict__ tgt,     // [B,SEQ]
                               float* __restrict__ loss_out) {
  __shared__ float h[32 * 65];
  __shared__ float gh[32 * 193];            // also hosts logits tmp (ld=65)
  __shared__ __align__(16) float gi_l[32 * 192];
  __shared__ float red[256];
  const int tid  = threadIdx.x;
  const int wave = tid >> 5;
  const int base = blockIdx.x * 32;

  // Stage gi tile into LDS with the async-copy engine (tracked by ASYNCcnt).
  {
    const uint64_t gsrc = (uint64_t)(uintptr_t)(gi + (size_t)base * 192);
    const uint32_t lds0 = (uint32_t)(uintptr_t)(&gi_l[0]);
    for (int c = tid; c < (32 * 192 * 4) / 16; c += 256)   // 6 x b128 per lane
      async_g2l_b128(lds0 + c * 16, gsrc + (uint64_t)c * 16);
  }
  for (int i = tid; i < 32 * 64; i += 256)
    h[(i >> 6) * 65 + (i & 63)] = h0[(size_t)(base + (i >> 6)) * 64 + (i & 63)];
  wait_asynccnt0();
  float lloss = 0.f;
  __syncthreads();

  for (int k = 0; k < TT; ++k) {
    // gh = h @ Whh^T   (24 wmma tiles: 2 Mt x 12 Nt, 3 per wave)
    for (int tile = wave; tile < 24; tile += 8) {
      const int mt = tile & 1, nt = tile >> 1;
      v8f acc = {};
#pragma unroll
      for (int k0 = 0; k0 < 64; k0 += 32) {
        v16h a = a_frag_f32(h, 65, mt * 16, k0, false);
        v16h b = b_frag_f16(Whh, 64, nt * 16, k0);
        acc = wmma_f16(a, b, acc);
      }
      store_d_lds(gh, 193, mt * 16, nt * 16, acc);
    }
    __syncthreads();
    // gates (gi is k-invariant; served from LDS)
    for (int i = tid; i < 32 * 64; i += 256) {
      const int row = i >> 6, j = i & 63;
      const float* gir = &gi_l[row * 192];
      const float hr = gh[row * 193 + j]       + bhh[j];
      const float hz = gh[row * 193 + 64 + j]  + bhh[64 + j];
      const float hn = gh[row * 193 + 128 + j] + bhh[128 + j];
      const float r  = 1.f / (1.f + __expf(-(gir[j] + hr)));
      const float z  = 1.f / (1.f + __expf(-(gir[64 + j] + hz)));
      const float nn = tanhf(gir[128 + j] + r * hn);
      h[row * 65 + j] = (1.f - z) * nn + z * h[row * 65 + j];
    }
    __syncthreads();
    // tmp = relu(h) @ adw1^T   (8 tiles, 1 per wave) -> stored into gh (ld=65)
    {
      const int mt = wave & 1, nt = wave >> 1;
      v8f acc = {};
#pragma unroll
      for (int k0 = 0; k0 < 64; k0 += 32) {
        v16h a = a_frag_f32(h, 65, mt * 16, k0, true);   // relu fused into A
        v16h b = b_frag_f16(adw1, 64, nt * 16, k0);
        acc = wmma_f16(a, b, acc);
      }
      store_d_lds(gh, 65, mt * 16, nt * 16, acc);
    }
    __syncthreads();
    if (tid < 32) {
      const int row = tid;
      float sacc = adb2[0];
      for (int c = 0; c < 64; ++c)
        sacc += fmaxf(gh[row * 65 + c] + adb1[c], 0.f) * adw2[c];
      const int grow = base + row;
      const int t = grow >> 8, b = grow & 255;
      const int j = t + 1 + k;
      if (j < SEQ) {
        const float tg = tgt[b * SEQ + j];
        const float sp = fmaxf(sacc, 0.f) + log1pf(__expf(-fabsf(sacc)));
        lloss += sp - sacc * tg;
      }
    }
    __syncthreads();
  }
  red[tid] = lloss;
  __syncthreads();
  for (int off = 128; off > 0; off >>= 1) {
    if (tid < off) red[tid] += red[tid + off];
    __syncthreads();
  }
  if (tid == 0) atomicAdd(loss_out, red[0] * (1.f / 256.f));
}

// ---------------------------------------------------------------------------
extern "C" void kernel_launch(void* const* d_in, const int* in_sizes, int n_in,
                              void* d_out, int out_size, void* d_ws, size_t ws_size,
                              hipStream_t stream) {
  (void)in_sizes; (void)n_in; (void)out_size; (void)ws_size;
  const float* actions = (const float*)d_in[0];
  const float* rewards = (const float*)d_in[1];
  const float* P[42];
  for (int i = 0; i < 42; ++i) P[i] = (const float*)d_in[2 + i];
  enum { AE_W1 = 0, AE_B1, AE_W2, AE_B2, RE_W1, RE_B1, RE_W2, RE_B2,
         ENC_WIH, ENC_WHH, ENC_BIH, ENC_BHH, BE_W1, BE_B1, BE_W2, BE_B2,
         TE_W1, TE_B1, TE_W2, TE_B2, LL_W1, LL_B1, LL_W2, LL_B2,
         LLT_W1, LLT_B1, LLT_W2, LLT_B2, DEC_WIH, DEC_WHH, DEC_BIH, DEC_BHH,
         AD_W1, AD_B1, AD_W2, AD_B2, TOP_EMB, TOP_CS, TOP_W,
         BOT_EMB, BOT_CS, BOT_W };

  // --- workspace bump allocator (256B aligned) ---
  size_t off = 0;
  auto alloc = [&](size_t bytes) -> char* {
    off = (off + 255) & ~(size_t)255;
    char* p = (char*)d_ws + off;
    off += bytes;
    return p;
  };
  float* xs       = (float*)alloc((size_t)ROWS * 32  * 4);
  float* gi_enc   = (float*)alloc((size_t)ROWS * 192 * 4);
  float* hs       = (float*)alloc((size_t)ROWS * 64  * 4);
  float* hid      = (float*)alloc((size_t)ROWS * 64  * 4);
  float* bottom_z = (float*)alloc((size_t)ROWS * 128 * 4);
  float* top_z    = (float*)alloc((size_t)ROWS * 128 * 4);
  float* m_seq    = (float*)alloc((size_t)ROWS * 128 * 4);
  float* mt_seq   = (float*)alloc((size_t)ROWS * 128 * 4);
  float* dec_feat = (float*)alloc((size_t)ROWS * 32  * 4);
  float* catb     = (float*)alloc((size_t)ROWS * 160 * 4);
  float* dec_h0   = (float*)alloc((size_t)ROWS * 64  * 4);
  float* gi_dec   = (float*)alloc((size_t)ROWS * 192 * 4);
  float* tgtbuf   = (float*)alloc((size_t)BB * SEQ * 4);

  auto cvt = [&](const float* src, int n) -> _Float16* {
    _Float16* dst = (_Float16*)alloc((size_t)n * sizeof(_Float16));
    f32_to_f16_kernel<<<(n + 255) / 256, 256, 0, stream>>>(src, dst, n);
    return dst;
  };
  _Float16* enc_wih_h = cvt(P[ENC_WIH], 192 * 32);
  _Float16* enc_whh_h = cvt(P[ENC_WHH], 192 * 64);
  _Float16* be_w1_h   = cvt(P[BE_W1],   64 * 64);
  _Float16* be_w2_h   = cvt(P[BE_W2],   128 * 64);
  _Float16* te_w1_h   = cvt(P[TE_W1],   64 * 128);
  _Float16* te_w2_h   = cvt(P[TE_W2],   128 * 64);
  _Float16* ll_w1_h   = cvt(P[LL_W1],   32 * 128);
  _Float16* ll_w2_h   = cvt(P[LL_W2],   32 * 32);
  _Float16* llt_w1_h  = cvt(P[LLT_W1],  64 * 160);
  _Float16* llt_w2_h  = cvt(P[LLT_W2],  64 * 64);
  _Float16* dec_wih_h = cvt(P[DEC_WIH], 192 * 32);
  _Float16* dec_whh_h = cvt(P[DEC_WHH], 192 * 64);
  _Float16* ad_w1_h   = cvt(P[AD_W1],   64 * 64);

  // --- outputs: [dec_loss, vq_loss, m, m_t, bot_idx, top_idx] ---
  float* out      = (float*)d_out;
  float* dec_loss = out + 0;
  float* vq_loss  = out + 1;
  float* m_T      = out + 2;
  float* mt_T     = out + 2 + (size_t)ROWS * 128;
  int*   bot_i    = (int*)(out + 2 + 2 * (size_t)ROWS * 128);
  int*   top_i    = bot_i + (size_t)ROWS * SS;
  hipMemsetAsync(d_out, 0, 2 * sizeof(float), stream);

  auto gemm = [&](const float* X, const _Float16* W, const float* b, float* Y,
                  int M, int N, int K, int relu) {
    const int tiles = (M / 16) * (N / 32);       // 16x32 tile per wave
    gemm_bias_kernel<<<(tiles + 7) / 8, 256, 0, stream>>>(X, W, b, Y, M, N, K, relu);
  };

  // 1) front-end features + targets
  embed_kernel<<<ROWS / 256, 256, 0, stream>>>(
      actions, rewards, P[AE_W1], P[AE_B1], P[AE_W2], P[AE_B2],
      P[RE_W1], P[RE_B1], P[RE_W2], P[RE_B2], xs);
  tgt_kernel<<<(BB * SEQ + 255) / 256, 256, 0, stream>>>(actions, tgtbuf, BB * SEQ);

  // 2) encoder: hoisted input-gate GEMM, then LDS-resident recurrent scan
  gemm(xs, enc_wih_h, P[ENC_BIH], gi_enc, ROWS, 192, 32, 0);
  enc_gru_kernel<<<BB / 32, 256, 0, stream>>>(gi_enc, enc_whh_h, P[ENC_BHH], hs);

  // 3) bottom / top encoders
  gemm(hs,       be_w1_h, P[BE_B1], hid,      ROWS, 64,  64,  1);
  gemm(hid,      be_w2_h, P[BE_B2], bottom_z, ROWS, 128, 64,  0);
  gemm(bottom_z, te_w1_h, P[TE_B1], hid,      ROWS, 64,  128, 1);
  gemm(hid,      te_w2_h, P[TE_B2], top_z,    ROWS, 128, 64,  0);

  // 4) sequential EMA quantizers (one block per codebook)
  quantize_kernel<<<SS, 256, 0, stream>>>(top_z, P[TOP_EMB], P[TOP_CS], P[TOP_W],
                                          mt_T, mt_seq, top_i, vq_loss);
  quantize_kernel<<<SS, 256, 0, stream>>>(bottom_z, P[BOT_EMB], P[BOT_CS], P[BOT_W],
                                          m_T, m_seq, bot_i, vq_loss);

  // 5) decoder feature path
  gemm(m_seq, ll_w1_h, P[LL_B1], hid,      ROWS, 32, 128, 1);
  gemm(hid,   ll_w2_h, P[LL_B2], dec_feat, ROWS, 32, 32,  0);
  concat_kernel<<<(ROWS * 160 + 255) / 256, 256, 0, stream>>>(
      dec_feat, mt_seq, catb, ROWS * 160);
  gemm(catb, llt_w1_h, P[LLT_B1], hid,    ROWS, 64, 160, 1);
  gemm(hid,  llt_w2_h, P[LLT_B2], dec_h0, ROWS, 64, 64,  0);
  gemm(dec_feat, dec_wih_h, P[DEC_BIH], gi_dec, ROWS, 192, 32, 0);

  // 6) decoder scan + logit head + masked BCE
  dec_gru_kernel<<<ROWS / 32, 256, 0, stream>>>(
      gi_dec, dec_h0, dec_whh_h, P[DEC_BHH],
      ad_w1_h, P[AD_B1], P[AD_W2], P[AD_B2], tgtbuf, dec_loss);
}
